// VectorQuantizer_47012712022212
// MI455X (gfx1250) — matrix operand order repaired and checked by reading, not compiled, not run on hardware
//
#include <hip/hip_runtime.h>

typedef __attribute__((ext_vector_type(16))) __bf16 v16bf;
typedef __attribute__((ext_vector_type(8)))  float  v8f;

#define NUM_EMB 1024
#define CDIM    256
#define HW      1024          // 32*32 spatial positions per image
#define ROWS_PER_BLOCK 64
#define KCHUNKS (CDIM/32)     // 8
#define NCHUNK  64
#define NSUB    4             // 4 WMMA n-subtiles of 16 per n-chunk
#define TOTAL_ELEMS 8388608.0f  // 32768 rows * 256 ch

// ---------------- Kernel A: codebook prep ----------------
// e -> bf16 hi/lo split, half-norms, zero the loss accumulator.
__global__ void vq_prep(const float* __restrict__ emb,
                        unsigned short* __restrict__ e_hi_raw,
                        unsigned short* __restrict__ e_lo_raw,
                        float* __restrict__ hnorm,
                        float* __restrict__ loss_acc) {
    __bf16* e_hi = reinterpret_cast<__bf16*>(e_hi_raw);
    __bf16* e_lo = reinterpret_cast<__bf16*>(e_lo_raw);
    const int n = blockIdx.x;       // 1024 blocks
    const int c = threadIdx.x;      // 256 threads
    float v = emb[n * CDIM + c];
    __bf16 hi = (__bf16)v;
    __bf16 lo = (__bf16)(v - (float)hi);
    e_hi[n * CDIM + c] = hi;
    e_lo[n * CDIM + c] = lo;

    float sq = v * v;
    #pragma unroll
    for (int off = 16; off > 0; off >>= 1) sq += __shfl_xor(sq, off, 32);
    __shared__ float part[8];
    const int lane = c & 31, w = c >> 5;
    if (lane == 0) part[w] = sq;
    __syncthreads();
    if (c == 0) {
        float s = 0.f;
        #pragma unroll
        for (int i = 0; i < 8; ++i) s += part[i];
        hnorm[n] = 0.5f * s;
        if (n == 0) *loss_acc = 0.f;
    }
}

// ---------------- Kernel B: scores via WMMA + argmax + gather + loss ----------------
// Block = 64 threads (2 waves); each wave owns 32 rows (two 16-row m-tiles) so each
// B fragment pair feeds 6 WMMAs: A streams from LDS (~170 B/clk/WGP), B from the
// L2-resident bf16 codebook through WGP$ (~170 B/clk/WGP) — both paths under capacity.
__global__ __launch_bounds__(64) void vq_main(
    const float* __restrict__ z,
    const float* __restrict__ emb,
    const unsigned short* __restrict__ e_hi_raw,
    const unsigned short* __restrict__ e_lo_raw,
    const float* __restrict__ hnorm,
    float* __restrict__ out,
    float* __restrict__ loss_acc)
{
    const __bf16* e_hi = reinterpret_cast<const __bf16*>(e_hi_raw);
    const __bf16* e_lo = reinterpret_cast<const __bf16*>(e_lo_raw);

    __shared__ __align__(64) unsigned char smem[65536];  // 64 rows x 256 ch x (hi+lo) bf16, fragment-ordered

    const int tid    = threadIdx.x;
    const int lane   = tid & 31;
    const int w      = tid >> 5;                 // wave 0..1, owns rows 32w..32w+31
    const int b      = blockIdx.x;               // 512 blocks
    const int nb     = b >> 4;                   // image index (16 hw-tiles per image)
    const int hwbase = (b & 15) * ROWS_PER_BLOCK;

    // ---- Phase 1: stage z tile into LDS in WMMA A-fragment order (hi & lo bf16) ----
    // fragment block (mt, kc, term): 512 halves laid out [lane][j]; mt = 16-row m-tile 0..3
    {
        __bf16* frag = (__bf16*)smem;
        for (int e = tid; e < ROWS_PER_BLOCK * CDIM; e += 64) {
            const int r = e & (ROWS_PER_BLOCK - 1);   // row: consecutive threads -> consecutive hw (coalesced)
            const int c = e >> 6;                     // channel
            const float v = z[((size_t)(nb * CDIM + c)) * HW + hwbase + r];
            const __bf16 hi = (__bf16)v;
            const __bf16 lo = (__bf16)(v - (float)hi);
            // A fragment order used here (contiguous k, not the ISA table order):
            //   lane l holds m = l&15; k = (l>>4)*16 + j
            const int mt  = r >> 4;
            const int m   = r & 15;
            const int kc  = c >> 5;
            const int k   = c & 31;
            const int jhi = (k >= 16) ? 1 : 0;
            const int kk  = k - (jhi << 4);
            const int gg  = kk >> 3;
            const int j   = kk;
            const int l   = (jhi << 4) + m;
            const int base = ((mt * KCHUNKS + kc) * 2) * 512 + l * 16 + j;  // halves
            frag[base]       = hi;
            frag[base + 512] = lo;
        }
    }
    __syncthreads();

    // ---- Phase 2: per-wave 32-row x 1024-emb argmax via bf16 hi/lo WMMA ----
    float bestv0[8], bestv1[8];
    int   besti0[8], besti1[8];
    #pragma unroll
    for (int v = 0; v < 8; ++v) {
        bestv0[v] = -3.0e38f; besti0[v] = 0;
        bestv1[v] = -3.0e38f; besti1[v] = 0;
    }

    const int g  = lane >> 4;   // half-wave group
    const int ln = lane & 15;
    const int mt0 = 2 * w;      // this wave's two m-tiles
    const int mt1 = 2 * w + 1;

    for (int nc = 0; nc < NUM_EMB / NCHUNK; ++nc) {
        // keep the per-(nc,kc) LDS A reloads inside the loop (blocks LICM -> no spills)
        asm volatile("" ::: "memory");

        v8f acc0[NSUB], acc1[NSUB];
        #pragma unroll
        for (int t = 0; t < NSUB; ++t) {
            acc0[t] = (v8f){0.f, 0.f, 0.f, 0.f, 0.f, 0.f, 0.f, 0.f};
            acc1[t] = (v8f){0.f, 0.f, 0.f, 0.f, 0.f, 0.f, 0.f, 0.f};
        }

        // prefetch next n-chunk of the codebook stream
        if (nc + 1 < NUM_EMB / NCHUNK) {
            __builtin_prefetch(e_hi + (size_t)(nc + 1) * NCHUNK * CDIM + lane * 64, 0, 3);
            __builtin_prefetch(e_lo + (size_t)(nc + 1) * NCHUNK * CDIM + lane * 64, 0, 3);
        }

        for (int kc = 0; kc < KCHUNKS; ++kc) {
            const __bf16* fr0 = (const __bf16*)smem + ((mt0 * KCHUNKS + kc) * 2) * 512 + lane * 16;
            const __bf16* fr1 = (const __bf16*)smem + ((mt1 * KCHUNKS + kc) * 2) * 512 + lane * 16;
            const v16bf a_hi0 = *(const v16bf*)fr0;
            const v16bf a_lo0 = *(const v16bf*)(fr0 + 512);
            const v16bf a_hi1 = *(const v16bf*)fr1;
            const v16bf a_lo1 = *(const v16bf*)(fr1 + 512);
            #pragma unroll
            for (int t = 0; t < NSUB; ++t) {
                // B fragment order used here (contiguous k, same as A): n = l&15, k = (l>>4)*16 + j -> 32 contiguous bytes
                const int n  = nc * NCHUNK + t * 16 + ln;
                const int ko = kc * 32 + g * 16;
                const v16bf b_hi = *(const v16bf*)(e_hi + n * CDIM + ko);
                const v16bf b_lo = *(const v16bf*)(e_lo + n * CDIM + ko);
                acc0[t] = __builtin_amdgcn_wmma_f32_16x16x32_bf16(false, a_hi0, false, b_hi,
                                                                  (short)0, acc0[t], false, false);
                acc0[t] = __builtin_amdgcn_wmma_f32_16x16x32_bf16(false, a_hi0, false, b_lo,
                                                                  (short)0, acc0[t], false, false);
                acc0[t] = __builtin_amdgcn_wmma_f32_16x16x32_bf16(false, a_lo0, false, b_hi,
                                                                  (short)0, acc0[t], false, false);
                acc1[t] = __builtin_amdgcn_wmma_f32_16x16x32_bf16(false, a_hi1, false, b_hi,
                                                                  (short)0, acc1[t], false, false);
                acc1[t] = __builtin_amdgcn_wmma_f32_16x16x32_bf16(false, a_hi1, false, b_lo,
                                                                  (short)0, acc1[t], false, false);
                acc1[t] = __builtin_amdgcn_wmma_f32_16x16x32_bf16(false, a_lo1, false, b_hi,
                                                                  (short)0, acc1[t], false, false);
            }
        }

        // fold scores into running argmax:  argmin d  ==  argmax(z.e - 0.5*||e||^2)
        #pragma unroll
        for (int t = 0; t < NSUB; ++t) {
            const int   n  = nc * NCHUNK + t * 16 + ln;
            const float hn = hnorm[n];
            #pragma unroll
            for (int v = 0; v < 8; ++v) {
                const float v0 = acc0[t][v] - hn;   // C/D layout: m = v + 8*g, n = ln
                const float v1 = acc1[t][v] - hn;
                if (v0 > bestv0[v]) { bestv0[v] = v0; besti0[v] = n; }
                if (v1 > bestv1[v]) { bestv1[v] = v1; besti1[v] = n; }
            }
        }
    }

    // 16-lane butterfly: lanes of a half-wave share m, differ in n
    #pragma unroll
    for (int v = 0; v < 8; ++v) {
        float bv0 = bestv0[v]; int bi0 = besti0[v];
        float bv1 = bestv1[v]; int bi1 = besti1[v];
        #pragma unroll
        for (int off = 8; off >= 1; off >>= 1) {
            const float ov0 = __shfl_xor(bv0, off, 16);
            const int   oi0 = __shfl_xor(bi0, off, 16);
            const float ov1 = __shfl_xor(bv1, off, 16);
            const int   oi1 = __shfl_xor(bi1, off, 16);
            if (ov0 > bv0 || (ov0 == bv0 && oi0 < bi0)) { bv0 = ov0; bi0 = oi0; }
            if (ov1 > bv1 || (ov1 == bv1 && oi1 < bi1)) { bv1 = ov1; bi1 = oi1; }
        }
        besti0[v] = bi0;
        besti1[v] = bi1;
    }

    __syncthreads();                    // all waves done reading A fragments
    int* idx_lds = (int*)smem;          // reuse LDS for the 64 winning indices
    if (ln == 0) {
        #pragma unroll
        for (int v = 0; v < 8; ++v) {
            idx_lds[mt0 * 16 + g * 8 + v] = besti0[v];
            idx_lds[mt1 * 16 + g * 8 + v] = besti1[v];
        }
    }
    __syncthreads();

    // ---- Phase 3: gather quantized rows (coalesced in z-layout) + exact fp32 loss ----
    // thread owns one row; per-channel iteration keeps 32 consecutive hw per wave store.
    float lsum = 0.f;
    const int id = idx_lds[tid];
    for (int c = 0; c < CDIM; ++c) {
        const float  q = emb[id * CDIM + c];
        const size_t o = ((size_t)(nb * CDIM + c)) * HW + hwbase + tid;
        const float zl = z[o];
        out[o] = q;
        const float d = zl - q;
        lsum += d * d;
    }
    #pragma unroll
    for (int off = 16; off > 0; off >>= 1) lsum += __shfl_xor(lsum, off, 32);
    if (lane == 0) atomicAdd(loss_acc, lsum);
}

// ---------------- Kernel C: finalize scalars ----------------
__global__ void vq_finalize(const float* __restrict__ loss_acc, float* __restrict__ out) {
    if (threadIdx.x == 0 && blockIdx.x == 0) {
        const float vq = *loss_acc * (1.0f / TOTAL_ELEMS);
        out[8388608] = vq;          // vq_loss
        out[8388609] = 0.25f * vq;  // commit_loss (same mean, weighted)
    }
}

extern "C" void kernel_launch(void* const* d_in, const int* in_sizes, int n_in,
                              void* d_out, int out_size, void* d_ws, size_t ws_size,
                              hipStream_t stream) {
    (void)in_sizes; (void)n_in; (void)out_size; (void)ws_size;
    const float* z   = (const float*)d_in[0];   // (32,256,32,32) fp32
    const float* emb = (const float*)d_in[1];   // (1024,256) fp32
    float* out = (float*)d_out;

    char* ws = (char*)d_ws;
    unsigned short* e_hi  = (unsigned short*)(ws);                       // 512 KB
    unsigned short* e_lo  = (unsigned short*)(ws + 512 * 1024);         // 512 KB
    float*          hnorm = (float*)(ws + 1024 * 1024);                 // 4 KB
    float*          loss  = (float*)(ws + 1024 * 1024 + 4096);          // 4 B

    vq_prep<<<NUM_EMB, CDIM, 0, stream>>>(emb, e_hi, e_lo, hnorm, loss);
    vq_main<<<512, 64, 0, stream>>>(z, emb, e_hi, e_lo, hnorm, out, loss);
    vq_finalize<<<1, 1, 0, stream>>>(loss, out);
}
